// graphSAGESINGLE_45698452030208
// MI455X (gfx1250) — matrix operand order repaired and checked by reading, not compiled, not run on hardware
//
#include <hip/hip_runtime.h>

typedef __attribute__((ext_vector_type(16))) __bf16 v16bf;
typedef __attribute__((ext_vector_type(8)))  float  v8f;

union VBf { v16bf v; unsigned u[8]; };

// ---------------------------------------------------------------------------
// GEMM: C[M,N] = A[M,K] @ B[K,N] (+bias), fp32 A/C, B pre-split into packed
// bf16 hi/lo words laid out exactly as the WMMA B fragment wants them:
//   Bpk[(k/2)*Nb + n] = (bf16(B[k+1][n]) << 16) | bf16(B[k][n])
// bf16x3 split (a_hi*b_hi + a_lo*b_hi + a_hi*b_lo) recovers ~fp32 accuracy.
// Block = 128 threads = 4 waves; each wave owns a 16-row M-tile (block = 64
// rows) and loops the 4 N-tiles of a 64-column stripe, reusing its A fragment
// -> 12 WMMAs per wave per k-step, A split once, B split zero times here.
// OOB rows zero-padded in LDS; stores row/col guarded after all WMMAs so
// EXEC is all-ones through the matrix ops.
// ---------------------------------------------------------------------------
__global__ void __launch_bounds__(128)
hgs_gemm_bf16x3(const float* __restrict__ A, const unsigned* __restrict__ Bhi,
                const unsigned* __restrict__ Blo, const float* __restrict__ bias,
                float* __restrict__ C, int M, int K, int Nb, int ldc, int Nstore) {
  extern __shared__ float As[];               // [64][K]
  const int tid  = threadIdx.x;
  const int wave = tid >> 5;
  const int lane = tid & 31;
  const int m0   = blockIdx.x * 64;

  // cooperative b128 load of 64 x K A tile (zero-fill rows beyond M)
  const int kv = K >> 2;                      // float4s per row
  for (int idx = tid; idx < 64 * kv; idx += 128) {
    int r = idx / kv, c = idx - r * kv;
    float4 val = make_float4(0.f, 0.f, 0.f, 0.f);
    if (m0 + r < M) val = ((const float4*)A)[(long long)(m0 + r) * kv + c];
    ((float4*)As)[idx] = val;
  }
  __syncthreads();

  const int half  = lane >> 4;
  const int nlane = lane & 15;
  const int mloc  = wave * 16 + nlane;        // LDS row this lane reads (A: M = lane%16)

  v8f acc[4] = {};
  for (int k0 = 0; k0 < K; k0 += 32) {
    // A fragment: 16-bit A 16x32 layout (ISA 7.12.2), split once per k-step
    v16bf ah, al;
#pragma unroll
    for (int v = 0; v < 8; ++v) {
      int kb = k0 + ((v >> 2) << 4) + (half << 3) + ((v & 3) << 1);
      float f0 = As[mloc * K + kb];
      float f1 = As[mloc * K + kb + 1];
      __bf16 h0 = (__bf16)f0, h1 = (__bf16)f1;
      ah[2 * v] = h0;  ah[2 * v + 1] = h1;
      al[2 * v]     = (__bf16)(f0 - (float)h0);
      al[2 * v + 1] = (__bf16)(f1 - (float)h1);
    }
#pragma unroll
    for (int nt = 0; nt < 4; ++nt) {
      const int ncol = blockIdx.y * 64 + nt * 16 + nlane;
      VBf bh, bl;
#pragma unroll
      for (int v = 0; v < 8; ++v) {
        // B fragment: lanes 0-15 hold K k0..k0+7,k0+16..k0+23; lanes 16-31 hold k0+8..k0+15,k0+24..k0+31
        long long w = (long long)((k0 >> 1) + ((v >> 2) << 3) + (half << 2) + (v & 3)) * Nb + ncol;
        bh.u[v] = Bhi[w];
        bl.u[v] = Blo[w];
      }
      acc[nt] = __builtin_amdgcn_wmma_f32_16x16x32_bf16(false, ah, false, bh.v, (short)0, acc[nt], false, false);
      acc[nt] = __builtin_amdgcn_wmma_f32_16x16x32_bf16(false, al, false, bh.v, (short)0, acc[nt], false, false);
      acc[nt] = __builtin_amdgcn_wmma_f32_16x16x32_bf16(false, ah, false, bl.v, (short)0, acc[nt], false, false);
    }
  }

#pragma unroll
  for (int nt = 0; nt < 4; ++nt) {
    const int ncol = blockIdx.y * 64 + nt * 16 + nlane;
    if (ncol < Nstore) {
      float bv = bias ? bias[ncol] : 0.0f;
#pragma unroll
      for (int r = 0; r < 8; ++r) {
        int m = m0 + wave * 16 + (half << 3) + r;   // f32 C/D: M = r + 8*half
        if (m < M) C[(long long)m * ldc + ncol] = acc[nt][r] + bv;
      }
    }
  }
}

// pack a fp32 [K][N] weight into bf16 hi/lo words in WMMA B-fragment order
__global__ void hgs_pack_b(const float* __restrict__ B, int K, int N,
                           unsigned* __restrict__ hi, unsigned* __restrict__ lo) {
  int idx = blockIdx.x * blockDim.x + threadIdx.x;   // over (K/2)*N
  int total = (K >> 1) * N;
  if (idx >= total) return;
  int k2 = idx / N, n = idx - k2 * N;
  float f0 = B[(long long)(2 * k2) * N + n];
  float f1 = B[(long long)(2 * k2 + 1) * N + n];
  __bf16 h0 = (__bf16)f0, h1 = (__bf16)f1;
  __bf16 l0 = (__bf16)(f0 - (float)h0), l1 = (__bf16)(f1 - (float)h1);
  union { __bf16 b[2]; unsigned u; } ph, pl;
  ph.b[0] = h0; ph.b[1] = h1;
  pl.b[0] = l0; pl.b[1] = l1;
  hi[idx] = ph.u;
  lo[idx] = pl.u;
}

// per-edge degree count (float so we can reuse the buffer as 1/deg)
__global__ void hgs_count_deg(const int* __restrict__ dst, float* __restrict__ cnt, int E) {
  int e = blockIdx.x * blockDim.x + threadIdx.x;
  if (e < E) unsafeAtomicAdd(&cnt[dst[e]], 1.0f);
}

__global__ void hgs_invert(float* __restrict__ c, int n) {
  int i = blockIdx.x * blockDim.x + threadIdx.x;
  if (i < n) c[i] = 1.0f / fmaxf(c[i], 1.0f);
}

// scatter-add: out[dst, coff..coff+63] += feat[src, 0..63] * inv[dst]
// (mean normalization folded into the scatter so accumulators can be shared)
__global__ void hgs_scatter64(const float* __restrict__ feat, const int* __restrict__ src,
                              const int* __restrict__ dst, const float* __restrict__ inv,
                              float* __restrict__ out, long long total, int ostride, int coff) {
  long long gid = (long long)blockIdx.x * blockDim.x + threadIdx.x;
  if (gid >= total) return;
  int e = (int)(gid >> 6);
  int c = (int)(gid & 63);
  int s = src[e], d = dst[e];
  unsafeAtomicAdd(&out[(long long)d * ostride + coff + c],
                  feat[(long long)s * 64 + c] * inv[d]);
}

__global__ void hgs_relu_ip(float* __restrict__ h, int n) {
  int i = blockIdx.x * blockDim.x + threadIdx.x;
  if (i < n) h[i] = fmaxf(h[i], 0.0f);
}

// relu p1 in place AND copy it into X2 columns 192..255 (self-term block)
__global__ void hgs_relu_p_to_x2(float* __restrict__ hp, float* __restrict__ X2, int n) {
  int i = blockIdx.x * blockDim.x + threadIdx.x;
  if (i < n) {
    float v = fmaxf(hp[i], 0.0f);
    hp[i] = v;
    int row = i >> 6, col = i & 63;
    X2[(long long)row * 256 + 192 + col] = v;
  }
}

// Fold weights: wcomb_p = wr1_wr+wr1_ci+wr1_rh (+bias sum); build stacked,
// zero-padded layer-2 weight [256 x 384] and bias [384].
__global__ void hgs_prep_weights(const float* __restrict__ wr1_wr, const float* __restrict__ wr1_ci,
                                 const float* __restrict__ wr1_rh,
                                 const float* __restrict__ b1_wr, const float* __restrict__ b1_ci,
                                 const float* __restrict__ b1_rh,
                                 const float* __restrict__ wl2_wr, const float* __restrict__ wl2_ci,
                                 const float* __restrict__ wl2_rh,
                                 const float* __restrict__ wr2_wr, const float* __restrict__ wr2_ci,
                                 const float* __restrict__ wr2_rh,
                                 const float* __restrict__ b2_wr, const float* __restrict__ b2_ci,
                                 const float* __restrict__ b2_rh,
                                 float* __restrict__ wcomb_p, float* __restrict__ bcomb_p,
                                 float* __restrict__ W2s, float* __restrict__ b2s) {
  int idx = blockIdx.x * blockDim.x + threadIdx.x;
  if (idx < 128 * 64) wcomb_p[idx] = wr1_wr[idx] + wr1_ci[idx] + wr1_rh[idx];
  if (idx < 64)       bcomb_p[idx] = b1_wr[idx] + b1_ci[idx] + b1_rh[idx];
  if (idx < 256 * 384) {
    int k2 = idx / 384, n = idx - k2 * 384;
    int blk = k2 >> 6, k = k2 & 63;
    float v = 0.0f;
    if (n < 349) {
      long long o = (long long)k * 349 + n;
      v = (blk == 0) ? wl2_wr[o] : (blk == 1) ? wl2_ci[o] : (blk == 2) ? wl2_rh[o]
                                 : (wr2_wr[o] + wr2_ci[o] + wr2_rh[o]);
    }
    W2s[idx] = v;
  }
  if (idx < 384) b2s[idx] = (idx < 349) ? (b2_wr[idx] + b2_ci[idx] + b2_rh[idx]) : 0.0f;
}

// ---------------------------------------------------------------------------
extern "C" void kernel_launch(void* const* d_in, const int* in_sizes, int n_in,
                              void* d_out, int out_size, void* d_ws, size_t ws_size,
                              hipStream_t stream) {
  constexpr int NP = 100000, NA = 100000, NF = 50000;
  const float* x_paper  = (const float*)d_in[0];
  const float* x_author = (const float*)d_in[1];
  const float* x_field  = (const float*)d_in[2];
  const int* ei_wr = (const int*)d_in[3];
  const int* ei_rw = (const int*)d_in[4];
  const int* ei_ci = (const int*)d_in[5];
  const int* ei_ht = (const int*)d_in[6];
  const int* ei_rh = (const int*)d_in[7];
  const int E_wr = in_sizes[3] / 2, E_rw = in_sizes[4] / 2, E_ci = in_sizes[5] / 2,
            E_ht = in_sizes[6] / 2, E_rh = in_sizes[7] / 2;
  const float *wl1_wr = (const float*)d_in[8],  *wr1_wr = (const float*)d_in[9],  *b1_wr = (const float*)d_in[10];
  const float *wl1_rw = (const float*)d_in[11], *wr1_rw = (const float*)d_in[12], *b1_rw = (const float*)d_in[13];
  const float *wl1_ci = (const float*)d_in[14], *wr1_ci = (const float*)d_in[15], *b1_ci = (const float*)d_in[16];
  const float *wl1_ht = (const float*)d_in[17], *wr1_ht = (const float*)d_in[18], *b1_ht = (const float*)d_in[19];
  const float *wl1_rh = (const float*)d_in[20], *wr1_rh = (const float*)d_in[21], *b1_rh = (const float*)d_in[22];
  const float *wl2_wr = (const float*)d_in[23], *wr2_wr = (const float*)d_in[24], *b2_wr = (const float*)d_in[25];
  const float *wl2_ci = (const float*)d_in[26], *wr2_ci = (const float*)d_in[27], *b2_ci = (const float*)d_in[28];
  const float *wl2_rh = (const float*)d_in[29], *wr2_rh = (const float*)d_in[30], *b2_rh = (const float*)d_in[31];
  float* out = (float*)d_out;

  // workspace layout (32-bit words)
  float* ws = (float*)d_ws;
  size_t off = 0;
  float* h_p = ws + off; off += (size_t)NP * 64;     // paper hidden / acc
  float* h_a = ws + off; off += (size_t)NA * 64;     // author hidden / acc
  float* h_f = ws + off; off += (size_t)NF * 64;     // field hidden / acc
  float* z   = ws + off; off += (size_t)NP * 64;     // per-type transformed src
  float* X2  = ws + off; off += (size_t)NP * 256;    // layer-2 concat activations
  float* W2s = ws + off; off += 256 * 384;           // stacked layer-2 weights (fp32)
  float* b2s = ws + off; off += 384;
  float* wcp = ws + off; off += 128 * 64;            // combined wr1 for paper (fp32)
  float* bcp = ws + off; off += 64;
  float* cnt = ws + off; off += 450000;              // 5 inverse-degree arrays
  float* c_wr = cnt;            // dst = paper (100k)
  float* c_rw = cnt + 100000;   // dst = author (100k)
  float* c_ci = cnt + 200000;   // dst = paper (100k)
  float* c_ht = cnt + 300000;   // dst = field (50k)
  float* c_rh = cnt + 350000;   // dst = paper (100k)
  // packed bf16 hi/lo weights: 8x (128x64 -> 4096 words) + 1x (256x384 -> 49152)
  unsigned* pk = (unsigned*)(ws + off);
  size_t poff = 0;
  auto take = [&](size_t n) { unsigned* p = pk + poff; poff += n; return p; };
  unsigned *p_wcp_h  = take(4096), *p_wcp_l  = take(4096);
  unsigned *p_srw_h  = take(4096), *p_srw_l  = take(4096);   // wr1_rw (author self)
  unsigned *p_sht_h  = take(4096), *p_sht_l  = take(4096);   // wr1_ht (field self)
  unsigned *p_lwr_h  = take(4096), *p_lwr_l  = take(4096);
  unsigned *p_lrw_h  = take(4096), *p_lrw_l  = take(4096);
  unsigned *p_lci_h  = take(4096), *p_lci_l  = take(4096);
  unsigned *p_lht_h  = take(4096), *p_lht_l  = take(4096);
  unsigned *p_lrh_h  = take(4096), *p_lrh_l  = take(4096);
  unsigned *p_w2_h   = take(49152), *p_w2_l  = take(49152);

  // 0) zero counts + X2, fold fp32 weights, then bf16-split-pack every B operand
  hipMemsetAsync(cnt, 0, 450000 * sizeof(float), stream);
  hipMemsetAsync(X2, 0, (size_t)NP * 256 * sizeof(float), stream);
  hgs_prep_weights<<<(256 * 384 + 255) / 256, 256, 0, stream>>>(
      wr1_wr, wr1_ci, wr1_rh, b1_wr, b1_ci, b1_rh,
      wl2_wr, wl2_ci, wl2_rh, wr2_wr, wr2_ci, wr2_rh,
      b2_wr, b2_ci, b2_rh, wcp, bcp, W2s, b2s);
  auto pack = [&](const float* B, int K, int N, unsigned* hi, unsigned* lo) {
    int total = (K / 2) * N;
    hgs_pack_b<<<(total + 255) / 256, 256, 0, stream>>>(B, K, N, hi, lo);
  };
  pack(wcp,    128, 64, p_wcp_h, p_wcp_l);
  pack(wr1_rw, 128, 64, p_srw_h, p_srw_l);
  pack(wr1_ht, 128, 64, p_sht_h, p_sht_l);
  pack(wl1_wr, 128, 64, p_lwr_h, p_lwr_l);
  pack(wl1_rw, 128, 64, p_lrw_h, p_lrw_l);
  pack(wl1_ci, 128, 64, p_lci_h, p_lci_l);
  pack(wl1_ht, 128, 64, p_lht_h, p_lht_l);
  pack(wl1_rh, 128, 64, p_lrh_h, p_lrh_l);
  pack(W2s,    256, 384, p_w2_h, p_w2_l);

  // 1) degrees -> inverse degrees (shared by both layers)
  hgs_count_deg<<<(E_wr + 255) / 256, 256, 0, stream>>>(ei_wr + E_wr, c_wr, E_wr);
  hgs_count_deg<<<(E_rw + 255) / 256, 256, 0, stream>>>(ei_rw + E_rw, c_rw, E_rw);
  hgs_count_deg<<<(E_ci + 255) / 256, 256, 0, stream>>>(ei_ci + E_ci, c_ci, E_ci);
  hgs_count_deg<<<(E_ht + 255) / 256, 256, 0, stream>>>(ei_ht + E_ht, c_ht, E_ht);
  hgs_count_deg<<<(E_rh + 255) / 256, 256, 0, stream>>>(ei_rh + E_rh, c_rh, E_rh);
  hgs_invert<<<(450000 + 255) / 256, 256, 0, stream>>>(cnt, 450000);

  auto gemm = [&](const float* A, const unsigned* Bhi, const unsigned* Blo, const float* bias,
                  float* C, int M, int K, int Nb, int ldc, int Nstore) {
    dim3 grid((M + 63) / 64, Nb / 64);
    hgs_gemm_bf16x3<<<grid, 128, (size_t)64 * K * sizeof(float), stream>>>(
        A, Bhi, Blo, bias, C, M, K, Nb, ldc, Nstore);
  };

  // 2) layer-1 self terms (init accumulators), biases folded in
  gemm(x_paper,  p_wcp_h, p_wcp_l, bcp,   h_p, NP, 128, 64, 64, 64);
  gemm(x_author, p_srw_h, p_srw_l, b1_rw, h_a, NA, 128, 64, 64, 64);
  gemm(x_field,  p_sht_h, p_sht_l, b1_ht, h_f, NF, 128, 64, 64, 64);

  // 3) per edge type: z = x_src @ wl1_t  then scatter z[src]/deg[dst] into h_dst
  auto edge1 = [&](const float* xsrc, int Nsrc, const unsigned* wh, const unsigned* wl,
                   const int* ei, int E, const float* inv, float* hdst) {
    gemm(xsrc, wh, wl, nullptr, z, Nsrc, 128, 64, 64, 64);
    long long total = (long long)E * 64;
    hgs_scatter64<<<(unsigned)((total + 255) / 256), 256, 0, stream>>>(
        z, ei, ei + E, inv, hdst, total, 64, 0);
  };
  edge1(x_author, NA, p_lwr_h, p_lwr_l, ei_wr, E_wr, c_wr, h_p);
  edge1(x_paper,  NP, p_lrw_h, p_lrw_l, ei_rw, E_rw, c_rw, h_a);
  edge1(x_paper,  NP, p_lci_h, p_lci_l, ei_ci, E_ci, c_ci, h_p);
  edge1(x_paper,  NP, p_lht_h, p_lht_l, ei_ht, E_ht, c_ht, h_f);
  edge1(x_field,  NF, p_lrh_h, p_lrh_l, ei_rh, E_rh, c_rh, h_p);

  // 4) ReLU (p1 also copied into X2 self-term block, cols 192..255)
  hgs_relu_p_to_x2<<<(NP * 64 + 255) / 256, 256, 0, stream>>>(h_p, X2, NP * 64);
  hgs_relu_ip<<<(NA * 64 + 255) / 256, 256, 0, stream>>>(h_a, NA * 64);
  hgs_relu_ip<<<(NF * 64 + 255) / 256, 256, 0, stream>>>(h_f, NF * 64);

  // 5) layer-2 aggregations into X2 column blocks (normalization folded in)
  {
    long long t;
    t = (long long)E_wr * 64;
    hgs_scatter64<<<(unsigned)((t + 255) / 256), 256, 0, stream>>>(h_a, ei_wr, ei_wr + E_wr, c_wr, X2, t, 256, 0);
    t = (long long)E_ci * 64;
    hgs_scatter64<<<(unsigned)((t + 255) / 256), 256, 0, stream>>>(h_p, ei_ci, ei_ci + E_ci, c_ci, X2, t, 256, 64);
    t = (long long)E_rh * 64;
    hgs_scatter64<<<(unsigned)((t + 255) / 256), 256, 0, stream>>>(h_f, ei_rh, ei_rh + E_rh, c_rh, X2, t, 256, 128);
  }

  // 6) single fused layer-2 GEMM: out[100k,349] = X2[100k,256] @ W2s[256,384] + b2s
  gemm(X2, p_w2_h, p_w2_l, b2s, out, NP, 256, 384, 349, 349);
}
